// SparseMoEBlock_9328668967102
// MI455X (gfx1250) — matrix operand order repaired and checked by reading, not compiled, not run on hardware
//
#include <hip/hip_runtime.h>
#include <cstdint>

// Problem constants (B=2, S=4096, D=1024, E=16, K=2, M=SM=2048)
#define T_TOK 8192
#define DHID  1024
#define NEXP  16
#define TOPK  2
#define MDIM  2048
#define CAPR  18432   // T*K + E*128 worst-case padded gather capacity

#if defined(__has_builtin)
#if __has_builtin(__builtin_amdgcn_global_load_async_to_lds_b128) && \
    __has_builtin(__builtin_amdgcn_s_wait_asynccnt)
#define USE_ASYNC 1
#endif
#endif
#ifndef USE_ASYNC
#define USE_ASYNC 0
#endif

typedef __attribute__((ext_vector_type(16))) __bf16 v16bf;
typedef __attribute__((ext_vector_type(8)))  float  v8f;
typedef __attribute__((ext_vector_type(4)))  int    v4i_t;

#if USE_ASYNC
typedef __attribute__((address_space(1))) v4i_t gv4i;   // global int4
typedef __attribute__((address_space(3))) v4i_t lv4i;   // LDS int4
__device__ __forceinline__ void async_cp16(const void* g, void* l) {
  __builtin_amdgcn_global_load_async_to_lds_b128((gv4i*)g, (lv4i*)l, 0, 0);
}
__device__ __forceinline__ void async_wait0() { __builtin_amdgcn_s_wait_asynccnt(0); }
#else
__device__ __forceinline__ void async_wait0() {}
#endif

__device__ __forceinline__ unsigned short f2bf(float f) {
  union { float f; unsigned int u; } v; v.f = f;
  unsigned int u = v.u;
  return (unsigned short)((u + 0x7FFFu + ((u >> 16) & 1u)) >> 16);  // RNE
}

// ---------------------------------------------------------------- converts
__global__ void cvt_bf16_kernel(const float* __restrict__ in,
                                unsigned short* __restrict__ out, long long n) {
  long long i = (long long)blockIdx.x * blockDim.x + threadIdx.x;
  long long stride = (long long)gridDim.x * blockDim.x;
  for (; i < n; i += stride) out[i] = f2bf(in[i]);
}

// ---------------------------------------------------------------- init
__global__ void init0_kernel(int* token_ids, float* wts, int* counts) {
  int i = blockIdx.x * blockDim.x + threadIdx.x;
  if (i < CAPR) { token_ids[i] = 0; wts[i] = 0.f; }
  if (i < NEXP) counts[i] = 0;
}

// ---------------------------------------------------------------- gate
__global__ void gate_kernel(const float* __restrict__ x, const float* __restrict__ gw,
                            int* __restrict__ tki, float* __restrict__ tkw,
                            int* __restrict__ counts) {
  int wave = threadIdx.x >> 5, lane = threadIdx.x & 31;
  int t = blockIdx.x * 8 + wave;
  float acc[NEXP];
#pragma unroll
  for (int e = 0; e < NEXP; ++e) acc[e] = 0.f;
  const float* xr = x + (size_t)t * DHID;
  for (int d0 = 0; d0 < DHID; d0 += 32) {
    float xv = xr[d0 + lane];
#pragma unroll
    for (int e = 0; e < NEXP; ++e) acc[e] += xv * gw[e * DHID + d0 + lane];
  }
#pragma unroll
  for (int e = 0; e < NEXP; ++e) {
#pragma unroll
    for (int off = 16; off > 0; off >>= 1) acc[e] += __shfl_xor(acc[e], off, 32);
  }
  if (lane == 0) {
    float m = acc[0];
    for (int e = 1; e < NEXP; ++e) m = fmaxf(m, acc[e]);
    float s = 0.f, p[NEXP];
    for (int e = 0; e < NEXP; ++e) { p[e] = __expf(acc[e] - m); s += p[e]; }
    float inv = 1.f / s;
    for (int e = 0; e < NEXP; ++e) p[e] *= inv;
    int i1 = 0; float v1 = p[0];
    for (int e = 1; e < NEXP; ++e) if (p[e] > v1) { v1 = p[e]; i1 = e; }
    int i2 = (i1 == 0) ? 1 : 0; float v2 = p[i2];
    for (int e = 0; e < NEXP; ++e) if (e != i1 && p[e] > v2) { v2 = p[e]; i2 = e; }
    tki[t * 2] = i1; tki[t * 2 + 1] = i2;
    tkw[t * 2] = v1; tkw[t * 2 + 1] = v2;
    atomicAdd(&counts[i1], 1);
    atomicAdd(&counts[i2], 1);
  }
}

// ---------------------------------------------------------------- scan (padded)
__global__ void scan_kernel(const int* counts, int* poff, int* cursors) {
  if (threadIdx.x == 0 && blockIdx.x == 0) {
    int run = 0;
    for (int e = 0; e < NEXP; ++e) {
      poff[e] = run; cursors[e] = run;
      run += (counts[e] + 127) & ~127;
    }
    poff[NEXP] = run;
  }
}

// ---------------------------------------------------------------- scatter
__global__ void scatter_kernel(const int* __restrict__ tki, const float* __restrict__ tkw,
                               int* cursors, int* token_ids, float* wts) {
  int t = blockIdx.x * blockDim.x + threadIdx.x;
  if (t >= T_TOK) return;
  for (int k = 0; k < TOPK; ++k) {
    int e = tki[t * 2 + k];
    int pos = atomicAdd(&cursors[e], 1);
    token_ids[pos] = t;
    wts[pos] = tkw[t * 2 + k];
  }
}

// ---------------------------------------------------------------- WMMA fragment loads
// A 16x32 bf16: lanes 0-15 -> K0..7 & K16..23 of row (lane&15); lanes 16-31 K8..15 & K24..31.
__device__ __forceinline__ v16bf load_frag_a(const unsigned short* base, int lane) {
  int r = lane & 15, hi = (lane >> 4) & 1;
  union { v16bf v; uint4 q[2]; } u;
  u.q[0] = *(const uint4*)(base + r * 32 + hi * 8);
  u.q[1] = *(const uint4*)(base + r * 32 + 16 + hi * 8);
  return u.v;
}
// B 32x16 bf16: lanes 0-15 hold K0..7 & K16..23 of column (lane&15); lanes 16-31 K8..15 & K24..31.
__device__ __forceinline__ v16bf load_frag_b(const unsigned short* base, int lane) {
  int c = lane & 15, hi = (lane >> 4) & 1;
  union { v16bf v; uint4 q[2]; } u;
  u.q[0] = *(const uint4*)(base + c * 32 + hi * 8);
  u.q[1] = *(const uint4*)(base + c * 32 + 16 + hi * 8);
  return u.v;
}

#define WMMA_BF16(A, B, C) \
  __builtin_amdgcn_wmma_f32_16x16x32_bf16(false, (A), false, (B), (short)0, (C), false, false)

// ---------------------------------------------------------------- up-proj
// 128x128 block tile, 8 waves; each wave computes 32x64 (2 A-frags x 4 B-frags)
// so every B fragment load feeds 4 WMMAs (2 rows x {gate,up}).
__global__ __launch_bounds__(256)
void mlp_up_kernel(const unsigned short* __restrict__ xb,
                   const unsigned short* __restrict__ Wg,
                   const unsigned short* __restrict__ Wu,
                   unsigned short* __restrict__ H,
                   const int* __restrict__ token_ids,   // null => identity gather
                   const float* __restrict__ wts,       // null => 1.0
                   const int* __restrict__ counts,      // null => full T
                   const int* __restrict__ poff) {
  const int KD = DHID, ND = MDIM;
  int e = blockIdx.z;
  int rowBase;
  if (counts) {
    int cap = (counts[e] + 127) & ~127;
    if ((int)blockIdx.x * 128 >= cap) return;
    rowBase = poff[e] + blockIdx.x * 128;
  } else {
    rowBase = blockIdx.x * 128;
  }
  int colBase = blockIdx.y * 128;
  const unsigned short* Wge = Wg + (size_t)e * KD * ND;
  const unsigned short* Wue = Wu + (size_t)e * KD * ND;

  __shared__ unsigned short sA[2][128 * 32];
  __shared__ unsigned short sBg[2][128 * 32];
  __shared__ unsigned short sBu[2][128 * 32];
  __shared__ int   sTok[128];
  __shared__ float sWt[128];

  int tid = threadIdx.x, wave = tid >> 5, lane = tid & 31;
  int wr = wave >> 1, wc = wave & 1;    // wave tile: rows wr*32, cols wc*64

  for (int i = tid; i < 128; i += 256) {
    sTok[i] = token_ids ? token_ids[rowBase + i] : (rowBase + i);
    sWt[i]  = wts ? wts[rowBase + i] : 1.f;
  }
  __syncthreads();

  // A staging: 512 16B chunks, 2 per thread. B staging: each thread stages a
  // pair of adjacent k-rows (bkp, bkp+1) for 8 columns -> packed b32 LDS stores.
  const int ar0 = tid >> 2,         ac0 = tid & 3;
  const int ar1 = (tid + 256) >> 2, ac1 = (tid + 256) & 3;
  const int bkp = (tid >> 4) * 2,   bc = tid & 15;
  const int tok0 = sTok[ar0], tok1 = sTok[ar1];

#if !USE_ASYNC
  uint4 ra0, ra1;
#endif
  uint4 rg0, rg1, ru0, ru1;
  auto ld = [&](int k0, int nb) {
#if USE_ASYNC
    async_cp16(xb + (size_t)tok0 * KD + k0 + ac0 * 8, &sA[nb][ar0 * 32 + ac0 * 8]);
    async_cp16(xb + (size_t)tok1 * KD + k0 + ac1 * 8, &sA[nb][ar1 * 32 + ac1 * 8]);
#else
    (void)nb;
    ra0 = *(const uint4*)(xb + (size_t)tok0 * KD + k0 + ac0 * 8);
    ra1 = *(const uint4*)(xb + (size_t)tok1 * KD + k0 + ac1 * 8);
#endif
    rg0 = *(const uint4*)(Wge + (size_t)(k0 + bkp) * ND + colBase + bc * 8);
    rg1 = *(const uint4*)(Wge + (size_t)(k0 + bkp + 1) * ND + colBase + bc * 8);
    ru0 = *(const uint4*)(Wue + (size_t)(k0 + bkp) * ND + colBase + bc * 8);
    ru1 = *(const uint4*)(Wue + (size_t)(k0 + bkp + 1) * ND + colBase + bc * 8);
  };
  auto st = [&](int b) {
#if !USE_ASYNC
    *(uint4*)&sA[b][ar0 * 32 + ac0 * 8] = ra0;
    *(uint4*)&sA[b][ar1 * 32 + ac1 * 8] = ra1;
#endif
    const unsigned short* hg0 = (const unsigned short*)&rg0;
    const unsigned short* hg1 = (const unsigned short*)&rg1;
    const unsigned short* hu0 = (const unsigned short*)&ru0;
    const unsigned short* hu1 = (const unsigned short*)&ru1;
#pragma unroll
    for (int j = 0; j < 8; ++j) {
      *(unsigned int*)&sBg[b][(bc * 8 + j) * 32 + bkp] =
          (unsigned int)hg0[j] | ((unsigned int)hg1[j] << 16);
      *(unsigned int*)&sBu[b][(bc * 8 + j) * 32 + bkp] =
          (unsigned int)hu0[j] | ((unsigned int)hu1[j] << 16);
    }
  };

  v8f accG[2][4], accU[2][4];
  v8f z = {};
#pragma unroll
  for (int ai = 0; ai < 2; ++ai)
#pragma unroll
    for (int f = 0; f < 4; ++f) { accG[ai][f] = z; accU[ai][f] = z; }

  ld(0, 0);
  st(0);
  async_wait0();
  __syncthreads();

  for (int k0 = 0, it = 0; k0 < KD; k0 += 32, ++it) {
    int cur = it & 1;
    bool more = (k0 + 32) < KD;
    if (more) ld(k0 + 32, cur ^ 1);

    v16bf a0 = load_frag_a(&sA[cur][(wr * 32) * 32], lane);
    v16bf a1 = load_frag_a(&sA[cur][(wr * 32 + 16) * 32], lane);
    v16bf bg = load_frag_b(&sBg[cur][(wc * 4) * 512], lane);
    v16bf bu = load_frag_b(&sBu[cur][(wc * 4) * 512], lane);
#pragma unroll
    for (int f = 0; f < 4; ++f) {
      v16bf bgn = bg, bun = bu;
      if (f < 3) {
        bgn = load_frag_b(&sBg[cur][(wc * 4 + f + 1) * 512], lane);
        bun = load_frag_b(&sBu[cur][(wc * 4 + f + 1) * 512], lane);
      }
      accG[0][f] = WMMA_BF16(a0, bg, accG[0][f]);
      accG[1][f] = WMMA_BF16(a1, bg, accG[1][f]);
      accU[0][f] = WMMA_BF16(a0, bu, accU[0][f]);
      accU[1][f] = WMMA_BF16(a1, bu, accU[1][f]);
      bg = bgn; bu = bun;
    }

    if (more) st(cur ^ 1);
    async_wait0();
    __syncthreads();
  }

  // epilogue: h = silu(g)*u*w, bf16 store. C frag: M = r + 8*(lane>=16), N = lane&15.
  int hi = (lane >> 4) & 1, ln = lane & 15;
#pragma unroll
  for (int ai = 0; ai < 2; ++ai) {
#pragma unroll
    for (int f = 0; f < 4; ++f) {
#pragma unroll
      for (int r = 0; r < 8; ++r) {
        int row = wr * 32 + ai * 16 + r + hi * 8;
        int col = colBase + wc * 64 + f * 16 + ln;
        float g = accG[ai][f][r], u = accU[ai][f][r];
        float hval = (g / (1.f + __expf(-g))) * u * sWt[row];
        H[(size_t)(rowBase + row) * ND + col] = f2bf(hval);
      }
    }
  }
}

// ---------------------------------------------------------------- down-proj: Y (+)= H*Wd
__global__ __launch_bounds__(256)
void mlp_down_kernel(const unsigned short* __restrict__ Hsrc,
                     const unsigned short* __restrict__ Wd,
                     float* __restrict__ y,
                     const int* __restrict__ token_ids,  // null => identity scatter
                     const int* __restrict__ counts,     // null => full T
                     const int* __restrict__ poff,
                     int do_atomic) {
  const int KD = MDIM, ND = DHID;
  int e = blockIdx.z;
  int rowBase;
  if (counts) {
    int cap = (counts[e] + 127) & ~127;
    if ((int)blockIdx.x * 128 >= cap) return;
    rowBase = poff[e] + blockIdx.x * 128;
  } else {
    rowBase = blockIdx.x * 128;
  }
  int colBase = blockIdx.y * 128;
  const unsigned short* Wde = Wd + (size_t)e * KD * ND;

  __shared__ unsigned short sA[2][128 * 32];
  __shared__ unsigned short sB[2][128 * 32];

  int tid = threadIdx.x, wave = tid >> 5, lane = tid & 31;
  int wr = wave >> 1, wc = wave & 1;

  const int ar0 = tid >> 2,         ac0 = tid & 3;
  const int ar1 = (tid + 256) >> 2, ac1 = (tid + 256) & 3;
  const int bkp = (tid >> 4) * 2,   bc = tid & 15;

#if !USE_ASYNC
  uint4 ra0, ra1;
#endif
  uint4 rb0, rb1;
  auto ld = [&](int k0, int nb) {
#if USE_ASYNC
    async_cp16(Hsrc + (size_t)(rowBase + ar0) * KD + k0 + ac0 * 8, &sA[nb][ar0 * 32 + ac0 * 8]);
    async_cp16(Hsrc + (size_t)(rowBase + ar1) * KD + k0 + ac1 * 8, &sA[nb][ar1 * 32 + ac1 * 8]);
#else
    (void)nb;
    ra0 = *(const uint4*)(Hsrc + (size_t)(rowBase + ar0) * KD + k0 + ac0 * 8);
    ra1 = *(const uint4*)(Hsrc + (size_t)(rowBase + ar1) * KD + k0 + ac1 * 8);
#endif
    rb0 = *(const uint4*)(Wde + (size_t)(k0 + bkp) * ND + colBase + bc * 8);
    rb1 = *(const uint4*)(Wde + (size_t)(k0 + bkp + 1) * ND + colBase + bc * 8);
  };
  auto st = [&](int b) {
#if !USE_ASYNC
    *(uint4*)&sA[b][ar0 * 32 + ac0 * 8] = ra0;
    *(uint4*)&sA[b][ar1 * 32 + ac1 * 8] = ra1;
#endif
    const unsigned short* h0 = (const unsigned short*)&rb0;
    const unsigned short* h1 = (const unsigned short*)&rb1;
#pragma unroll
    for (int j = 0; j < 8; ++j) {
      *(unsigned int*)&sB[b][(bc * 8 + j) * 32 + bkp] =
          (unsigned int)h0[j] | ((unsigned int)h1[j] << 16);
    }
  };

  v8f acc[2][4]; v8f z = {};
#pragma unroll
  for (int ai = 0; ai < 2; ++ai)
#pragma unroll
    for (int f = 0; f < 4; ++f) acc[ai][f] = z;

  ld(0, 0);
  st(0);
  async_wait0();
  __syncthreads();

  for (int k0 = 0, it = 0; k0 < KD; k0 += 32, ++it) {
    int cur = it & 1;
    bool more = (k0 + 32) < KD;
    if (more) ld(k0 + 32, cur ^ 1);

    v16bf a0 = load_frag_a(&sA[cur][(wr * 32) * 32], lane);
    v16bf a1 = load_frag_a(&sA[cur][(wr * 32 + 16) * 32], lane);
    v16bf b = load_frag_b(&sB[cur][(wc * 4) * 512], lane);
#pragma unroll
    for (int f = 0; f < 4; ++f) {
      v16bf bn = b;
      if (f < 3) bn = load_frag_b(&sB[cur][(wc * 4 + f + 1) * 512], lane);
      acc[0][f] = WMMA_BF16(a0, b, acc[0][f]);
      acc[1][f] = WMMA_BF16(a1, b, acc[1][f]);
      b = bn;
    }

    if (more) st(cur ^ 1);
    async_wait0();
    __syncthreads();
  }

  int hi = (lane >> 4) & 1, ln = lane & 15;
#pragma unroll
  for (int ai = 0; ai < 2; ++ai) {
#pragma unroll
    for (int f = 0; f < 4; ++f) {
#pragma unroll
      for (int r = 0; r < 8; ++r) {
        int row = wr * 32 + ai * 16 + r + hi * 8;
        int col = colBase + wc * 64 + f * 16 + ln;
        int tok = token_ids ? token_ids[rowBase + row] : (rowBase + row);
        float v = acc[ai][f][r];
        if (do_atomic) atomicAdd(&y[(size_t)tok * ND + col], v);  // padded rows add 0.0
        else y[(size_t)tok * ND + col] = v;
      }
    }
  }
}

// ---------------------------------------------------------------- launch
extern "C" void kernel_launch(void* const* d_in, const int* in_sizes, int n_in,
                              void* d_out, int out_size, void* d_ws, size_t ws_size,
                              hipStream_t stream) {
  const float* x  = (const float*)d_in[0];
  const float* gw = (const float*)d_in[1];
  const float* wg = (const float*)d_in[2];
  const float* wu = (const float*)d_in[3];
  const float* wd = (const float*)d_in[4];
  const float* sg = (const float*)d_in[5];
  const float* su = (const float*)d_in[6];
  const float* sd = (const float*)d_in[7];
  float* y = (float*)d_out;
  (void)in_sizes; (void)n_in; (void)out_size;

  char* p = (char*)d_ws;
  auto alloc = [&](size_t bytes) { char* r = p; p += (bytes + 255) & ~(size_t)255; return r; };
  unsigned short* xb  = (unsigned short*)alloc((size_t)T_TOK * DHID * 2);
  unsigned short* wgb = (unsigned short*)alloc((size_t)NEXP * DHID * MDIM * 2);
  unsigned short* wub = (unsigned short*)alloc((size_t)NEXP * DHID * MDIM * 2);
  unsigned short* wdb = (unsigned short*)alloc((size_t)NEXP * MDIM * DHID * 2);
  unsigned short* sgb = (unsigned short*)alloc((size_t)DHID * MDIM * 2);
  unsigned short* sub = (unsigned short*)alloc((size_t)DHID * MDIM * 2);
  unsigned short* sdb = (unsigned short*)alloc((size_t)MDIM * DHID * 2);
  unsigned short* Hr  = (unsigned short*)alloc((size_t)CAPR * MDIM * 2);
  unsigned short* Hs  = (unsigned short*)alloc((size_t)T_TOK * MDIM * 2);
  int*   tki  = (int*)alloc((size_t)T_TOK * 2 * 4);
  float* tkw  = (float*)alloc((size_t)T_TOK * 2 * 4);
  int*   tids = (int*)alloc((size_t)CAPR * 4);
  float* twts = (float*)alloc((size_t)CAPR * 4);
  int* counts  = (int*)alloc(256);
  int* poff    = (int*)alloc(256);
  int* cursors = (int*)alloc(256);
  if ((size_t)(p - (char*)d_ws) > ws_size) return;  // workspace too small

  auto cvt = [&](const float* src, unsigned short* dst, long long n) {
    cvt_bf16_kernel<<<2048, 256, 0, stream>>>(src, dst, n);
  };
  cvt(x,  xb,  (long long)T_TOK * DHID);
  cvt(wg, wgb, (long long)NEXP * DHID * MDIM);
  cvt(wu, wub, (long long)NEXP * DHID * MDIM);
  cvt(wd, wdb, (long long)NEXP * MDIM * DHID);
  cvt(sg, sgb, (long long)DHID * MDIM);
  cvt(su, sub, (long long)DHID * MDIM);
  cvt(sd, sdb, (long long)MDIM * DHID);

  // routing
  init0_kernel<<<(CAPR + 255) / 256, 256, 0, stream>>>(tids, twts, counts);
  gate_kernel<<<T_TOK / 8, 256, 0, stream>>>(x, gw, tki, tkw, counts);
  scan_kernel<<<1, 32, 0, stream>>>(counts, poff, cursors);
  scatter_kernel<<<T_TOK / 256, 256, 0, stream>>>(tki, tkw, cursors, tids, twts);

  // shared expert first (plain stores initialize y everywhere)
  mlp_up_kernel<<<dim3(T_TOK / 128, MDIM / 128, 1), 256, 0, stream>>>(
      xb, sgb, sub, Hs, nullptr, nullptr, nullptr, nullptr);
  mlp_down_kernel<<<dim3(T_TOK / 128, DHID / 128, 1), 256, 0, stream>>>(
      Hs, sdb, y, nullptr, nullptr, nullptr, 0);

  // routed experts (grouped GEMM over gathered tokens, atomic combine)
  mlp_up_kernel<<<dim3(T_TOK / 128, MDIM / 128, NEXP), 256, 0, stream>>>(
      xb, wgb, wub, Hr, tids, twts, counts, poff);
  mlp_down_kernel<<<dim3(T_TOK / 128, DHID / 128, NEXP), 256, 0, stream>>>(
      Hr, wdb, y, tids, counts, poff, 1);
}
